// FeedForwardNetwork_s2_89979564851595
// MI455X (gfx1250) — compile-verified, matrix operand order checked by reading, not run on hardware
//
#include <hip/hip_runtime.h>
#include <math.h>

// ---------------------------------------------------------------------------
// SO3 FFN (s2 grid) fully fused per-token kernel for gfx1250 (MI455X).
// All GEMMs run on v_wmma_f32_16x16x32_f16 (f32 accumulate), intermediates in
// LDS (~224KB of the 320KB WGP LDS, dynamic alloc), zero HBM intermediates.
// Masked stores use address-redirect (cndmask) instead of EXEC-mask branches.
// ---------------------------------------------------------------------------

typedef __attribute__((ext_vector_type(16))) _Float16 v16h;
typedef __attribute__((ext_vector_type(8)))  _Float16 v8h;
typedef __attribute__((ext_vector_type(8)))  float    v8f;

#define P_PAD 352          // 324 grid points padded to 22 tiles of 16
#define LDS_BYTES 229920   // bufA + bufB + swgt + h1t + sH2 + sgate + dump

__device__ __forceinline__ v8f wmma_f16(v16h a, v16h b, v8f c) {
  // D = A(16x32 f16) * B(32x16 f16) + C(16x16 f32)
  return __builtin_amdgcn_wmma_f32_16x16x32_f16(false, a, false, b, (short)0, c,
                                                false, false);
}

// Load one 16x32 f16 fragment in the CDNA5 A-matrix lane layout from a
// row-major [rows][ldk] source. The identical layout serves the B operand when
// the source is stored as [N][K] row-major (i.e. W^T), per ISA 7.12.2:
//   lane<16 : K = k0+0..7  and k0+16..23 of row (lane&15)
//   lane>=16: K = k0+8..15 and k0+24..31 of row (lane&15)
__device__ __forceinline__ v16h load_frag(const _Float16* __restrict__ src,
                                          int row0, int ldk, int k0, int lane) {
  const _Float16* p = src + (row0 + (lane & 15)) * ldk + k0 + 8 * (lane >> 4);
  v8h lo = *(const v8h*)p;
  v8h hi = *(const v8h*)(p + 16);
  return __builtin_shufflevector(lo, hi, 0, 1, 2, 3, 4, 5, 6, 7,
                                         8, 9, 10, 11, 12, 13, 14, 15);
}

__device__ __forceinline__ float silu_f(float x) {
  return x / (1.0f + __expf(-x));
}

// ---------------------------------------------------------------------------
// Prep: f32 -> f16 weight transposes + padded to_grid / from_grid matrices.
// ---------------------------------------------------------------------------
__global__ void prep_kernel(const float* __restrict__ w1, const float* __restrict__ w2,
                            const float* __restrict__ wg1, const float* __restrict__ wg2,
                            const float* __restrict__ wg3,
                            const float* __restrict__ tg, const float* __restrict__ fg,
                            _Float16* __restrict__ w1t, _Float16* __restrict__ w2t,
                            _Float16* __restrict__ wgt,
                            _Float16* __restrict__ tgA, _Float16* __restrict__ fgA) {
  const int stride = gridDim.x * blockDim.x;
  const int tid0 = blockIdx.x * blockDim.x + threadIdx.x;
  // w1t/w2t[l][d][c] = w[l][c][d]
  for (int idx = tid0; idx < 5 * 128 * 128; idx += stride) {
    int l = idx >> 14, rem = idx & 16383, d = rem >> 7, c = rem & 127;
    w1t[idx] = (_Float16)w1[(l << 14) + (c << 7) + d];
    w2t[idx] = (_Float16)w2[(l << 14) + (c << 7) + d];
  }
  // wgt[k][d][c] = wgk[c][d]
  for (int idx = tid0; idx < 3 * 128 * 128; idx += stride) {
    int wsel = idx >> 14, rem = idx & 16383, d = rem >> 7, c = rem & 127;
    const float* w = (wsel == 0) ? wg1 : (wsel == 1) ? wg2 : wg3;
    wgt[idx] = (_Float16)w[(c << 7) + d];
  }
  // tgA[p][i] (352x32, zero padded), p = b*18+a flattened row-major
  for (int idx = tid0; idx < P_PAD * 32; idx += stride) {
    int p = idx >> 5, i = idx & 31;
    tgA[idx] = (p < 324 && i < 25) ? (_Float16)tg[p * 25 + i] : (_Float16)0.0f;
  }
  // fgA[i][p] (32x352, zero padded)
  for (int idx = tid0; idx < 32 * P_PAD; idx += stride) {
    int i = idx / P_PAD, p = idx % P_PAD;
    fgA[idx] = (p < 324 && i < 25) ? (_Float16)fg[p * 25 + i] : (_Float16)0.0f;
  }
}

// ---------------------------------------------------------------------------
// Fused kernel: one workgroup (8 wave32) per token n.
// ---------------------------------------------------------------------------
__global__ void __launch_bounds__(256, 1)
so3_ffn_fused(const float* __restrict__ x,
              const float* __restrict__ wsw, const float* __restrict__ bs,
              const float* __restrict__ b1, const float* __restrict__ b2,
              const _Float16* __restrict__ w1t, const _Float16* __restrict__ w2t,
              const _Float16* __restrict__ wgt,
              const _Float16* __restrict__ tgA, const _Float16* __restrict__ fgA,
              float* __restrict__ out, float* __restrict__ gdump) {
  extern __shared__ char smem[];
  _Float16* bufA  = (_Float16*)smem;          // [352][128] grid activations
  _Float16* bufB  = bufA + P_PAD * 128;       // [352][128], layer3 reuses as g3t[128][352]
  _Float16* swgt  = bufB + P_PAD * 128;       // [128][128] current MLP weight (W^T)
  _Float16* h1t   = swgt + 128 * 128;         // [128][32]  h1 transposed (c, i)
  _Float16* sH2   = h1t + 128 * 32;           // [32][128]  from-grid result (i, c)
  float*    sgate = (float*)(sH2 + 32 * 128); // [128]
  _Float16* sdump = (_Float16*)(sgate + 128); // 16 halves: garbage sink
  _Float16* xh    = bufA;                     // [32][128]  overlaps bufA (dead after stage A)

  const int tid = threadIdx.x;
  const int lane = tid & 31;
  const int wv = tid >> 5;                    // 8 waves
  const int n = blockIdx.x;
  const float* xn = x + (size_t)n * 3200;     // x[n] : [25][128]

  // ---- phase 1: stage x[n] -> f16 LDS (rows 25..31 zero), zero h1t pad rows,
  //      compute gating = silu(x0 @ ws + bs) with plain VALU (tiny) ----
  for (int idx = tid; idx < 32 * 128; idx += 256)
    xh[idx] = (idx < 3200) ? (_Float16)xn[idx] : (_Float16)0.0f;
  for (int idx = tid; idx < 128 * 7; idx += 256) {
    int c = idx / 7, i = 25 + idx % 7;
    h1t[c * 32 + i] = (_Float16)0.0f;
  }
  if (tid < 128) {
    float a = bs[tid];
    for (int k = 0; k < 128; ++k) a = fmaf(xn[k], wsw[k * 128 + tid], a);
    sgate[tid] = silu_f(a);
  }
  __syncthreads();

  // ---- stage A: h1 = so3_linear(x, w1, b1); stored transposed h1t[c][i] ----
  // per degree l: D[m=i_local][nn=c] = sum_k xh[lsq+m][k] * w1t[l][c][k]
  for (int t = wv; t < 40; t += 8) {
    int l = t >> 3, ct = t & 7;
    int lsq = l * l, lend = (l + 1) * (l + 1);
    const _Float16* wl = w1t + l * 16384;
    v8f acc = {};
#pragma unroll
    for (int k = 0; k < 4; ++k) {
      v16h a = load_frag(xh, lsq, 128, k * 32, lane);
      v16h b = load_frag(wl, ct * 16, 128, k * 32, lane);
      acc = wmma_f16(a, b, acc);
    }
    int c = ct * 16 + (lane & 15);
    float bias0 = b1[c];
#pragma unroll
    for (int r = 0; r < 8; ++r) {
      int m = r + 8 * (lane >> 4), i = lsq + m;
      float v = acc[r] + (i == 0 ? bias0 : 0.0f);
      _Float16* dst = (i < lend) ? (h1t + c * 32 + i) : sdump;  // redirect, no branch
      *dst = (_Float16)v;
    }
  }
  __syncthreads();

  // ---- to_grid: bufA[p][c] = sum_i tg[p][i] * h1[i][c]
  //      D[m=c][nn=p] = h1t[c][i] * tgA[p][i]^T; transposed packed store ----
  for (int t = wv; t < 176; t += 8) {
    int mt = t & 7, pt = t >> 3;  // 8 c-tiles x 22 p-tiles
    v16h a = load_frag(h1t, mt * 16, 32, 0, lane);
    v16h b = load_frag(tgA, pt * 16, 32, 0, lane);
    v8f acc = {};
    acc = wmma_f16(a, b, acc);
    v8h h;
#pragma unroll
    for (int r = 0; r < 8; ++r) h[r] = (_Float16)acc[r];
    *(v8h*)(bufA + (pt * 16 + (lane & 15)) * 128 + mt * 16 + 8 * (lane >> 4)) = h;
  }
  __syncthreads();

  // ---- grid MLP: 3 layers of [352][128] @ [128][128] ----
  // D[m=c_out][nn=p] = W^T[c_out][c_in] * g[p][c_in]^T
  for (int layer = 0; layer < 3; ++layer) {
    const _Float16* wsrc = wgt + layer * 16384;
    for (int idx = tid; idx < 2048; idx += 256)
      ((uint4*)swgt)[idx] = ((const uint4*)wsrc)[idx];
    __syncthreads();
    const _Float16* inb = (layer == 1) ? bufB : bufA;
    _Float16* outb = (layer == 1) ? bufA : bufB;
    for (int pt = wv; pt < 22; pt += 8) {
      v16h bfr[4];
#pragma unroll
      for (int k = 0; k < 4; ++k)
        bfr[k] = load_frag(inb, pt * 16, 128, k * 32, lane);
      for (int mt = 0; mt < 8; ++mt) {
        v8f acc = {};
#pragma unroll
        for (int k = 0; k < 4; ++k) {
          v16h a = load_frag(swgt, mt * 16, 128, k * 32, lane);
          acc = wmma_f16(a, bfr[k], acc);
        }
        if (layer < 2) {  // silu + transposed packed store -> row-major [p][c]
          v8h h;
#pragma unroll
          for (int r = 0; r < 8; ++r) h[r] = (_Float16)silu_f(acc[r]);
          *(v8h*)(outb + (pt * 16 + (lane & 15)) * 128 + mt * 16 + 8 * (lane >> 4)) = h;
        } else {          // layer3: store as g3t[c_out][p] (scatter, no act)
#pragma unroll
          for (int r = 0; r < 8; ++r)
            outb[(mt * 16 + r + 8 * (lane >> 4)) * P_PAD + pt * 16 + (lane & 15)] =
                (_Float16)acc[r];
        }
      }
    }
    __syncthreads();
  }

  // ---- from_grid: sH2[i][c] = sum_p fg[p][i] * g3[p][c]
  //      D[m=c][nn=i] = g3t[c][p] * fgA[i][p]^T; transposed packed store ----
  const _Float16* g3t = bufB;  // [128][352]
  for (int t = wv; t < 16; t += 8) {
    int mt = t & 7, nt = t >> 3;  // 8 c-tiles x 2 i-tiles
    v8f acc = {};
#pragma unroll
    for (int k = 0; k < 11; ++k) {
      v16h a = load_frag(g3t, mt * 16, P_PAD, k * 32, lane);
      v16h b = load_frag(fgA, nt * 16, P_PAD, k * 32, lane);
      acc = wmma_f16(a, b, acc);
    }
    v8h h;
#pragma unroll
    for (int r = 0; r < 8; ++r) h[r] = (_Float16)acc[r];
    *(v8h*)(sH2 + (nt * 16 + (lane & 15)) * 128 + mt * 16 + 8 * (lane >> 4)) = h;
  }
  __syncthreads();
  if (tid < 128) sH2[tid] = (_Float16)sgate[tid];  // replace l=0 row with gating
  __syncthreads();

  // ---- stage C: out[n] = so3_linear(h2, w2, b2), straight to global f32 ----
  for (int t = wv; t < 40; t += 8) {
    int l = t >> 3, ct = t & 7;
    int lsq = l * l, lend = (l + 1) * (l + 1);
    const _Float16* wl = w2t + l * 16384;
    v8f acc = {};
#pragma unroll
    for (int k = 0; k < 4; ++k) {
      v16h a = load_frag(sH2, lsq, 128, k * 32, lane);
      v16h b = load_frag(wl, ct * 16, 128, k * 32, lane);
      acc = wmma_f16(a, b, acc);
    }
    int c = ct * 16 + (lane & 15);
    float bias0 = b2[c];
#pragma unroll
    for (int r = 0; r < 8; ++r) {
      int m = r + 8 * (lane >> 4), i = lsq + m;
      float v = acc[r] + (i == 0 ? bias0 : 0.0f);
      float* dst = (i < lend) ? (out + ((size_t)n * 25 + i) * 128 + c) : gdump;
      *dst = v;  // redirect invalid rows to dump, no EXEC branching
    }
  }
}

// ---------------------------------------------------------------------------
extern "C" void kernel_launch(void* const* d_in, const int* in_sizes, int n_in,
                              void* d_out, int out_size, void* d_ws, size_t ws_size,
                              hipStream_t stream) {
  const float* x   = (const float*)d_in[0];
  const float* w1  = (const float*)d_in[1];
  const float* b1  = (const float*)d_in[2];
  const float* wsw = (const float*)d_in[3];
  const float* bs  = (const float*)d_in[4];
  const float* wg1 = (const float*)d_in[5];
  const float* wg2 = (const float*)d_in[6];
  const float* wg3 = (const float*)d_in[7];
  const float* w2  = (const float*)d_in[8];
  const float* b2  = (const float*)d_in[9];
  const float* tg  = (const float*)d_in[10];
  const float* fg  = (const float*)d_in[11];
  float* out = (float*)d_out;

  // workspace: f16 transposed weights + padded grid matrices (~471 KB) + dump
  _Float16* w1t = (_Float16*)d_ws;
  _Float16* w2t = w1t + 5 * 128 * 128;
  _Float16* wgt = w2t + 5 * 128 * 128;
  _Float16* tgA = wgt + 3 * 128 * 128;
  _Float16* fgA = tgA + P_PAD * 32;
  float*    gdump = (float*)(fgA + 32 * P_PAD);  // 16-float garbage sink

  prep_kernel<<<128, 256, 0, stream>>>(w1, w2, wg1, wg2, wg3, tg, fg,
                                       w1t, w2t, wgt, tgA, fgA);
  so3_ffn_fused<<<2048, 256, LDS_BYTES, stream>>>(x, wsw, bs, b1, b2,
                                                  w1t, w2t, wgt, tgA, fgA,
                                                  out, gdump);
}
